// ChunkedSSM_91319594648397
// MI455X (gfx1250) — matrix operand order repaired and checked by reading, not compiled, not run on hardware
//
#include <hip/hip_runtime.h>
#include <hip/hip_bf16.h>

// ---------------- problem constants (from reference) ----------------
#define DIMX    2048
#define STATE   64
#define DTRANK  128
#define BATCH   2
#define SEQ     4096
#define BLROWS  (BATCH*SEQ)          // 8192
#define PDIM    (DTRANK + 2*STATE)   // 256
#define CHUNK   256
#define NCHUNK  (SEQ/CHUNK)          // 16

typedef __attribute__((ext_vector_type(16))) __bf16 v16bf;
typedef __attribute__((ext_vector_type(8)))  __bf16 v8bf;
typedef __attribute__((ext_vector_type(8)))  float  v8f;

__device__ __forceinline__ float fast_exp2(float x) {
    return __builtin_amdgcn_exp2f(x);   // v_exp_f32
}

// A fragment (16x32 bf16, lane l = row l&15, hi = l>>4):
// elems 0..7  <-> k = 8*hi + 0..7      (contiguous 16B)
// elems 8..15 <-> k = 16 + 8*hi + 0..7 (contiguous 16B)
__device__ __forceinline__ v16bf load_a_frag(const __bf16* p) {
    v8bf lo = *(const v8bf*)p;
    v8bf hi = *(const v8bf*)(p + 16);
    return __builtin_shufflevector(lo, hi, 0, 1, 2, 3, 4, 5, 6, 7,
                                           8, 9, 10, 11, 12, 13, 14, 15);
}

// ---------------- small prep kernels ----------------
__global__ __launch_bounds__(256) void cvt_bf16_kernel(const float* __restrict__ s,
                                                       __bf16* __restrict__ d, int n) {
    int i = blockIdx.x * 256 + threadIdx.x;
    if (i < n) d[i] = (__bf16)s[i];
}

__global__ __launch_bounds__(256) void a2_kernel(const float* __restrict__ Alog,
                                                 float* __restrict__ A2, int n) {
    int i = blockIdx.x * 256 + threadIdx.x;
    if (i < n) A2[i] = -__expf(Alog[i]) * 1.44269504088896f;  // A * log2(e)
}

// ---------------- LayerNorm: one block per row of 2048; writes f32 + bf16 ----------
__global__ __launch_bounds__(256) void layernorm_kernel(const float* __restrict__ x,
                                                        const float* __restrict__ w,
                                                        const float* __restrict__ b,
                                                        float* __restrict__ xn,
                                                        __bf16* __restrict__ xnb) {
    const int row = blockIdx.x;
    const float* xr = x   + (size_t)row * DIMX;
    float*       xo = xn  + (size_t)row * DIMX;
    __bf16*      xb = xnb + (size_t)row * DIMX;
    float vals[8];
    float s = 0.f, s2 = 0.f;
#pragma unroll
    for (int i = 0; i < 8; i++) {
        float v = xr[threadIdx.x + i * 256];
        vals[i] = v; s += v; s2 += v * v;
    }
#pragma unroll
    for (int off = 16; off > 0; off >>= 1) {
        s  += __shfl_xor(s,  off, 32);
        s2 += __shfl_xor(s2, off, 32);
    }
    __shared__ float red[2][8];
    int wave = threadIdx.x >> 5, lane = threadIdx.x & 31;
    if (lane == 0) { red[0][wave] = s; red[1][wave] = s2; }
    __syncthreads();
    s = 0.f; s2 = 0.f;
#pragma unroll
    for (int i = 0; i < 8; i++) { s += red[0][i]; s2 += red[1][i]; }
    float mu  = s * (1.f / DIMX);
    float var = s2 * (1.f / DIMX) - mu * mu;
    float rs  = rsqrtf(var + 1e-5f);
#pragma unroll
    for (int i = 0; i < 8; i++) {
        int c = threadIdx.x + i * 256;
        float v = (vals[i] - mu) * rs * w[c] + b[c];
        xo[c] = v;
        xb[c] = (__bf16)v;
    }
}

// ---------------- bf16 WMMA GEMM:  C[m,n] = sum_k A[m,k] * W[n,k] ----------------
// A: bf16 row-major (lda). W: bf16 row-major (N,K).
// Block tile 128x128, 8 waves (2M x 4N), wave tile 64x32 (8 WMMA accumulators).
// mode 1: fused  v = softplus(v + bias[col]).  Cbf != nullptr: also store bf16 copy.
__global__ __launch_bounds__(256) void gemm_wmma_bf16(const __bf16* __restrict__ A, int lda,
                                                      const __bf16* __restrict__ W, int K,
                                                      float* __restrict__ C, int ldc,
                                                      __bf16* __restrict__ Cbf,
                                                      const float* __restrict__ bias, int mode) {
    const int lane  = threadIdx.x & 31;
    const int wave  = threadIdx.x >> 5;
    const int lrow  = lane & 15;
    const int hi    = lane >> 4;
    const int waveM = wave >> 2;      // 0..1
    const int waveN = wave & 3;       // 0..3
    const int tileM = blockIdx.y * 128 + waveM * 64;
    const int tileN = blockIdx.x * 128 + waveN * 32;

    v8f acc[4][2];
#pragma unroll
    for (int mi = 0; mi < 4; mi++)
#pragma unroll
        for (int ni = 0; ni < 2; ni++)
#pragma unroll
            for (int e = 0; e < 8; e++) acc[mi][ni][e] = 0.f;

    for (int kk = 0; kk < K; kk += 32) {
        v16bf af[4];
#pragma unroll
        for (int mi = 0; mi < 4; mi++)
            af[mi] = load_a_frag(A + (size_t)(tileM + mi * 16 + lrow) * lda + kk + hi * 8);
        // B fragment: lane l holds col (l&15); same k order as A fragment
        v16bf bfrag[2];
#pragma unroll
        for (int ni = 0; ni < 2; ni++)
            bfrag[ni] = load_a_frag(W + (size_t)(tileN + ni * 16 + lrow) * K + kk + hi * 8);
#pragma unroll
        for (int mi = 0; mi < 4; mi++)
#pragma unroll
            for (int ni = 0; ni < 2; ni++)
                acc[mi][ni] = __builtin_amdgcn_wmma_f32_16x16x32_bf16(
                    false, af[mi], false, bfrag[ni], (short)0, acc[mi][ni], false, false);
    }
    // D layout: vgpr r -> row r + 8*hi, col = lrow
#pragma unroll
    for (int mi = 0; mi < 4; mi++)
#pragma unroll
        for (int ni = 0; ni < 2; ni++)
#pragma unroll
            for (int r = 0; r < 8; r++) {
                int row = tileM + mi * 16 + r + 8 * hi;
                int col = tileN + ni * 16 + lrow;
                float v = acc[mi][ni][r];
                if (mode == 1) {
                    v += bias[col];
                    float sp = __logf(1.f + __expf(v));   // softplus
                    v = (v > 20.f) ? v : sp;
                }
                C[(size_t)row * ldc + col] = v;
                if (Cbf) Cbf[(size_t)row * ldc + col] = (__bf16)v;
            }
}

// ---------------- scan pass 1: per-chunk local state (h starts at 0) ----------------
__global__ __launch_bounds__(256) void scan_pass1(const float* __restrict__ delta,
                                                  const float* __restrict__ proj,
                                                  const float* __restrict__ xn,
                                                  const float* __restrict__ A2,
                                                  float* __restrict__ Hend,
                                                  float* __restrict__ dtsum) {
    const int d = blockIdx.x * 256 + threadIdx.x;
    const int c = blockIdx.y;
    const int b = blockIdx.z;
    float A2r[STATE], h[STATE];
#pragma unroll
    for (int n = 0; n < STATE; n++) { A2r[n] = A2[(size_t)d * STATE + n]; h[n] = 0.f; }
    float dts = 0.f;
    __shared__ float sB[16 * STATE];
    const size_t rowbase = (size_t)b * SEQ;
    for (int ls = 0; ls < CHUNK; ls += 16) {
        const int t0 = c * CHUNK + ls;
        __syncthreads();
        for (int q = threadIdx.x; q < 16 * STATE; q += 256)
            sB[q] = proj[(rowbase + t0 + (q >> 6)) * PDIM + DTRANK + (q & 63)];
        __syncthreads();
        for (int i = 0; i < 16; i++) {
            const int t = t0 + i;
            float dt = delta[(rowbase + t) * DIMX + d];
            float xt = xn[(rowbase + t) * DIMX + d];
            dts += dt;
            float u = dt * xt;
            const float* Bp = &sB[i * STATE];
#pragma unroll
            for (int n = 0; n < STATE; n++) {
                float dA = fast_exp2(dt * A2r[n]);
                h[n] = __builtin_fmaf(dA, h[n], u * Bp[n]);
            }
        }
    }
    const size_t o = (((size_t)b * NCHUNK + c) * DIMX + d) * STATE;
#pragma unroll
    for (int n = 0; n < STATE; n++) Hend[o + n] = h[n];
    dtsum[((size_t)b * NCHUNK + c) * DIMX + d] = dts;
}

// ---------------- chunk combine: sequential over 16 chunks, parallel over (b,d,n) ----
__global__ __launch_bounds__(256) void scan_combine(const float* __restrict__ Hend,
                                                    const float* __restrict__ dtsum,
                                                    const float* __restrict__ A2,
                                                    float* __restrict__ Hin) {
    const int idx = blockIdx.x * 256 + threadIdx.x;   // B*DIMX*STATE = 262144
    const int n = idx & (STATE - 1);
    const int d = (idx >> 6) & (DIMX - 1);
    const int b = idx >> 17;
    const float a2 = A2[(size_t)d * STATE + n];
    float h = 0.f;
    for (int c = 0; c < NCHUNK; c++) {
        const size_t o = (((size_t)b * NCHUNK + c) * DIMX + d) * STATE + n;
        Hin[o] = h;
        float S = dtsum[((size_t)b * NCHUNK + c) * DIMX + d];
        h = __builtin_fmaf(fast_exp2(a2 * S), h, Hend[o]);   // Pi exp(dt*A) = exp2(A2*sum dt)
    }
}

// ---------------- scan pass 2: recompute with carry-in, produce z = y + x*D (bf16) ----
__global__ __launch_bounds__(256) void scan_pass2(const float* __restrict__ delta,
                                                  const float* __restrict__ proj,
                                                  const float* __restrict__ xn,
                                                  const float* __restrict__ xin,
                                                  const float* __restrict__ A2,
                                                  const float* __restrict__ Hin,
                                                  const float* __restrict__ Dp,
                                                  __bf16* __restrict__ zb) {
    const int d = blockIdx.x * 256 + threadIdx.x;
    const int c = blockIdx.y;
    const int b = blockIdx.z;
    float A2r[STATE], h[STATE];
    const size_t ho = (((size_t)b * NCHUNK + c) * DIMX + d) * STATE;
#pragma unroll
    for (int n = 0; n < STATE; n++) { A2r[n] = A2[(size_t)d * STATE + n]; h[n] = Hin[ho + n]; }
    const float Dpar = Dp[d];
    __shared__ float sBC[16 * 2 * STATE];   // per step: [0:64]=B_ssm, [64:128]=C_ssm
    const size_t rowbase = (size_t)b * SEQ;
    for (int ls = 0; ls < CHUNK; ls += 16) {
        const int t0 = c * CHUNK + ls;
        __syncthreads();
        for (int q = threadIdx.x; q < 16 * 2 * STATE; q += 256)
            sBC[q] = proj[(rowbase + t0 + (q >> 7)) * PDIM + DTRANK + (q & 127)];
        __syncthreads();
        for (int i = 0; i < 16; i++) {
            const int t = t0 + i;
            float dt = delta[(rowbase + t) * DIMX + d];
            float xt = xn[(rowbase + t) * DIMX + d];
            float u = dt * xt;
            const float* Bp = &sBC[i * 2 * STATE];
            const float* Cp = Bp + STATE;
            float y = 0.f;
#pragma unroll
            for (int n = 0; n < STATE; n++) {
                float dA = fast_exp2(dt * A2r[n]);
                h[n] = __builtin_fmaf(dA, h[n], u * Bp[n]);
                y = __builtin_fmaf(h[n], Cp[n], y);
            }
            float xo = xin[(rowbase + t) * DIMX + d];
            zb[(rowbase + t) * DIMX + d] = (__bf16)(y + xo * Dpar);
        }
    }
}

// ---------------- host-side orchestration ----------------
extern "C" void kernel_launch(void* const* d_in, const int* in_sizes, int n_in,
                              void* d_out, int out_size, void* d_ws, size_t ws_size,
                              hipStream_t stream) {
    const float* x       = (const float*)d_in[0];
    const float* norm_w  = (const float*)d_in[1];
    const float* norm_b  = (const float*)d_in[2];
    const float* xprojw  = (const float*)d_in[3];   // (256, 2048)
    const float* dtprojw = (const float*)d_in[4];   // (2048, 128)
    const float* dtprojb = (const float*)d_in[5];   // (2048,)
    const float* Alog    = (const float*)d_in[6];   // (2048, 64)
    const float* Dp      = (const float*)d_in[7];   // (2048,)
    const float* outw    = (const float*)d_in[8];   // (2048, 2048)
    float* out = (float*)d_out;

    char* ws = (char*)d_ws;
    size_t off = 0;
    auto alloc = [&](size_t bytes) -> char* {
        char* p = ws + off;
        off = (off + bytes + 255) & ~(size_t)255;
        return p;
    };
    float*  xn     = (float*) alloc((size_t)BLROWS * DIMX * 4);     // 64 MB (scan x_t)
    __bf16* xnb    = (__bf16*)alloc((size_t)BLROWS * DIMX * 2);     // 32 MB (GEMM A)
    float*  dl     = (float*) alloc((size_t)BLROWS * DIMX * 4);     // delta, 64 MB
    __bf16* zb     = (__bf16*)alloc((size_t)BLROWS * DIMX * 2);     // 32 MB (GEMM A)
    float*  proj   = (float*) alloc((size_t)BLROWS * PDIM * 4);     // 8 MB (scan B/C)
    __bf16* projb  = (__bf16*)alloc((size_t)BLROWS * PDIM * 2);     // 4 MB (GEMM A)
    __bf16* Wx     = (__bf16*)alloc((size_t)PDIM * DIMX * 2);
    __bf16* Wdt    = (__bf16*)alloc((size_t)DIMX * DTRANK * 2);
    __bf16* Wout   = (__bf16*)alloc((size_t)DIMX * DIMX * 2);
    float*  A2     = (float*) alloc((size_t)DIMX * STATE * 4);
    float*  dtsum  = (float*) alloc((size_t)BATCH * NCHUNK * DIMX * 4);
    float*  Hend   = (float*) alloc((size_t)BATCH * NCHUNK * DIMX * STATE * 4);  // 16 MB
    float*  Hin    = (float*) alloc((size_t)BATCH * NCHUNK * DIMX * STATE * 4);  // 16 MB

    // weight conversion + A2
    {
        int n1 = PDIM * DIMX;
        cvt_bf16_kernel<<<(n1 + 255) / 256, 256, 0, stream>>>(xprojw, Wx, n1);
        int n2 = DIMX * DTRANK;
        cvt_bf16_kernel<<<(n2 + 255) / 256, 256, 0, stream>>>(dtprojw, Wdt, n2);
        int n3 = DIMX * DIMX;
        cvt_bf16_kernel<<<(n3 + 255) / 256, 256, 0, stream>>>(outw, Wout, n3);
        int n4 = DIMX * STATE;
        a2_kernel<<<(n4 + 255) / 256, 256, 0, stream>>>(Alog, A2, n4);
    }
    // layernorm (f32 + bf16)
    layernorm_kernel<<<BLROWS, 256, 0, stream>>>(x, norm_w, norm_b, xn, xnb);
    // proj = xn @ x_proj_w^T   (M=8192, N=256, K=2048), f32 + bf16 outputs
    gemm_wmma_bf16<<<dim3(PDIM / 128, BLROWS / 128), 256, 0, stream>>>(
        xnb, DIMX, Wx, DIMX, proj, PDIM, projb, nullptr, 0);
    // delta = softplus(delta_r @ dt_proj_w^T + b)  (M=8192, N=2048, K=128; A = proj_bf16[:, :128])
    gemm_wmma_bf16<<<dim3(DIMX / 128, BLROWS / 128), 256, 0, stream>>>(
        projb, PDIM, Wdt, DTRANK, dl, DIMX, nullptr, dtprojb, 1);
    // chunked scan
    scan_pass1<<<dim3(DIMX / 256, NCHUNK, BATCH), 256, 0, stream>>>(
        dl, proj, xn, A2, Hend, dtsum);
    scan_combine<<<(BATCH * DIMX * STATE) / 256, 256, 0, stream>>>(Hend, dtsum, A2, Hin);
    scan_pass2<<<dim3(DIMX / 256, NCHUNK, BATCH), 256, 0, stream>>>(
        dl, proj, xn, x, A2, Hin, Dp, zb);
    // out = z @ out_proj_w^T   (M=8192, N=2048, K=2048)
    gemm_wmma_bf16<<<dim3(DIMX / 128, BLROWS / 128), 256, 0, stream>>>(
        zb, DIMX, Wout, DIMX, out, DIMX, nullptr, nullptr, 0);
    (void)in_sizes; (void)n_in; (void)out_size; (void)ws_size;
}
